// TransformerLayerAbs_62045097558237
// MI455X (gfx1250) — compile-verified, matrix operand order checked by reading, not run on hardware
//
#include <hip/hip_runtime.h>
#include <hip/hip_bf16.h>
#include <math.h>

// ---------------------------------------------------------------------------
// TransformerLayerAbs for MI455X (gfx1250, wave32, WMMA + TDM).
// Compute-bound (~620 GFLOP vs ~0.5GB traffic) -> all matmuls through
// v_wmma_f32_16x16x32_bf16; GEMM block tiles staged into LDS by the Tensor
// Data Mover (tensor_load_to_lds, TENSORcnt), double-buffered.
// abs_encoding folded into K and V streams:
//   scores = q @ (k+abs)^T / 8 ;  attn = softmax(scores) @ (v+abs)
// ---------------------------------------------------------------------------

typedef unsigned short u16;                                   // bf16 storage
typedef __attribute__((ext_vector_type(16))) __bf16 v16bf;    // WMMA A/B frag
typedef __attribute__((ext_vector_type(8)))  float  v8f;      // WMMA C/D frag
typedef __attribute__((ext_vector_type(4)))  unsigned int u32x4;
typedef __attribute__((ext_vector_type(4)))  int i32x4;
typedef __attribute__((ext_vector_type(8)))  int i32x8;

#define B_SZ   8
#define S_LEN  2048
#define H_DIM  1024
#define I_DIM  4096
#define MROWS  (B_SZ * S_LEN)   // 16384
#define SPAD   2056             // LDS score stride (multiple of 8 -> 16B aligned)

#if defined(__has_builtin)
#if __has_builtin(__builtin_amdgcn_tensor_load_to_lds)
#define HAVE_TDM 1
#endif
#endif

union FragU { v16bf v; uint4 q[2]; };

__device__ __forceinline__ u16 f2bf(float f) {
  union { float f; unsigned u; } v; v.f = f;
  unsigned r = v.u + 0x7FFFu + ((v.u >> 16) & 1u);   // round-to-nearest-even
  return (u16)(r >> 16);
}
__device__ __forceinline__ float bf2f(u16 h) {
  union { unsigned u; float f; } v; v.u = ((unsigned)h) << 16;
  return v.f;
}
// Load one 16x32 bf16 WMMA fragment (CDNA5 layout): lane holds row `lr`,
// elements e<8 at k0+hi*8+e, e>=8 at k0+16+hi*8+(e-8)  -> two b128 loads.
__device__ __forceinline__ void frag_ld(FragU& fr, const u16* p) {
  fr.q[0] = *(const uint4*)(p);
  fr.q[1] = *(const uint4*)(p + 16);
}

#ifdef HAVE_TDM
// TDM: DMA a rows x cols bf16 tile (row stride = tK elements) from global to
// LDS (contiguous tile-major). D# per CDNA5 ISA.
__device__ __forceinline__ void tdm_load_2d(unsigned lds_addr, const u16* gptr,
                                            int rows, int cols, int tK, int tM) {
  unsigned long long ga = (unsigned long long)gptr;
  u32x4 g0;
  g0[0] = 1u;                                            // count=1, user mode
  g0[1] = lds_addr;                                      // LDS byte address
  g0[2] = (unsigned)(ga & 0xFFFFFFFFu);                  // global_addr[31:0]
  g0[3] = (unsigned)((ga >> 32) & 0x1FFFFFFu) | (2u << 30); // ga[56:32], type=2
  i32x8 g1;
  g1[0] = (int)(1u << 16);                               // data_size=1 -> 2 bytes
  g1[1] = (int)(((unsigned)tK & 0xFFFFu) << 16);         // tensor_dim0[15:0]
  g1[2] = (int)((((unsigned)tK >> 16) & 0xFFFFu) |
                (((unsigned)tM & 0xFFFFu) << 16));       // td0[31:16] | td1[15:0]
  g1[3] = (int)((((unsigned)tM >> 16) & 0xFFFFu) |
                (((unsigned)cols & 0xFFFFu) << 16));     // td1[31:16] | tile_dim0
  g1[4] = (int)((unsigned)rows & 0xFFFFu);               // tile_dim1 (tile_dim2=0)
  g1[5] = (int)(unsigned)tK;                             // tensor_dim0_stride[31:0]
  g1[6] = 0;
  g1[7] = 0;
  i32x4 z4 = {0, 0, 0, 0};
#if __clang_major__ >= 23
  i32x8 z8 = {0, 0, 0, 0, 0, 0, 0, 0};
  __builtin_amdgcn_tensor_load_to_lds(g0, g1, z4, z4, z8, 0);
#else
  __builtin_amdgcn_tensor_load_to_lds(g0, g1, z4, z4, 0);
#endif
}
#endif

// -------------------------------- utility ----------------------------------
__global__ void cvt_bf16_kernel(const float* __restrict__ x, u16* __restrict__ y, long n) {
  long i = (long)blockIdx.x * blockDim.x + threadIdx.x;
  if (i < n) y[i] = f2bf(x[i]);
}

// W[K][N] f32 -> Wt[N][K] bf16 (one-time; makes B-fragments contiguous loads)
__global__ void transpose_cvt_kernel(const float* __restrict__ W, u16* __restrict__ Wt,
                                     int K, int N) {
  long idx = (long)blockIdx.x * blockDim.x + threadIdx.x;
  if (idx >= (long)K * N) return;
  int k = (int)(idx / N), n = (int)(idx % N);
  Wt[(long)n * K + k] = f2bf(W[idx]);
}

// -------------------------------- GEMM -------------------------------------
// C[M,N] = A[M,K](bf16) @ Wt[N,K](bf16)^T + bias, epilogue variants.
// Block: 256 thr = 8 waves; wave tile 32x32 (2x2 WMMA accs); block tile 128x64.
// Per k-step: wave0 DMAs A(128x32) and B(64x32) tiles into LDS via TDM
// (double-buffered, s_wait_tensorcnt + barrier); all waves eat ds_load_b128
// fragments and issue 4 WMMAs.
// epi: 0=bf16  1=bf16,+addv  2=bf16 transposed per-batch [b][n][s],+addv
//      3=f32   4=gelu->bf16
__global__ void gemm_bf16_wmma(const u16* __restrict__ A, const u16* __restrict__ Bt,
                               const float* __restrict__ bias,
                               const u16* __restrict__ addv, void* __restrict__ out,
                               int M, int N, int K, int epi) {
  __shared__ u16 sA[2][128 * 32];
  __shared__ u16 sB[2][64 * 32];
  const int lane = threadIdx.x & 31;
  const int w    = threadIdx.x >> 5;
  const int lr   = lane & 15, hi = lane >> 4;
  const int mblk = blockIdx.x * 128;
  const int nblk = blockIdx.y * 64;
  const int wm = (w & 3) * 32;          // wave row origin within block tile
  const int wn = (w >> 2) * 32;         // wave col origin within block tile
  const int nsteps = K >> 5;

#ifdef HAVE_TDM
  if (w == 0) {   // prologue: stage k-step 0
    tdm_load_2d((unsigned)(unsigned long long)(void*)&sA[0][0],
                A + (long)mblk * K, 128, 32, K, M);
    tdm_load_2d((unsigned)(unsigned long long)(void*)&sB[0][0],
                Bt + (long)nblk * K, 64, 32, K, N);
  }
#else
  { // cooperative fallback copy for k-step 0
    for (int t = threadIdx.x; t < 128 * 4; t += 256) {
      int r = t >> 2, c = (t & 3) * 8;
      *(uint4*)&sA[0][r * 32 + c] = *(const uint4*)(A + (long)(mblk + r) * K + c);
    }
    for (int t = threadIdx.x; t < 64 * 4; t += 256) {
      int r = t >> 2, c = (t & 3) * 8;
      *(uint4*)&sB[0][r * 32 + c] = *(const uint4*)(Bt + (long)(nblk + r) * K + c);
    }
  }
#endif

  v8f acc[2][2] = {};
  for (int s = 0; s < nsteps; ++s) {
    const int buf = s & 1;
#ifdef HAVE_TDM
    if (w == 0) {
      if (s + 1 < nsteps) {   // stage next k-step into the other buffer
        const int kk = (s + 1) << 5;
        tdm_load_2d((unsigned)(unsigned long long)(void*)&sA[buf ^ 1][0],
                    A + (long)mblk * K + kk, 128, 32, K, M);
        tdm_load_2d((unsigned)(unsigned long long)(void*)&sB[buf ^ 1][0],
                    Bt + (long)nblk * K + kk, 64, 32, K, N);
        __builtin_amdgcn_s_wait_tensorcnt(2);  // step-s tiles landed (in-order)
      } else {
        __builtin_amdgcn_s_wait_tensorcnt(0);
      }
    }
    __syncthreads();
#else
    __syncthreads();
#endif

    FragU a[2], b[2];
    #pragma unroll
    for (int i = 0; i < 2; ++i) {
      frag_ld(a[i], &sA[buf][(wm + i * 16 + lr) * 32 + hi * 8]);
      frag_ld(b[i], &sB[buf][(wn + i * 16 + lr) * 32 + hi * 8]);
    }
    #pragma unroll
    for (int i = 0; i < 2; ++i)
      #pragma unroll
      for (int j = 0; j < 2; ++j)
        acc[i][j] = __builtin_amdgcn_wmma_f32_16x16x32_bf16(
            false, a[i].v, false, b[j].v, (short)0, acc[i][j], false, false);

#ifndef HAVE_TDM
    __syncthreads();
    if (s + 1 < nsteps) {   // cooperative copy of next tile
      const int kk = (s + 1) << 5;
      for (int t = threadIdx.x; t < 128 * 4; t += 256) {
        int r = t >> 2, c = (t & 3) * 8;
        *(uint4*)&sA[buf ^ 1][r * 32 + c] = *(const uint4*)(A + (long)(mblk + r) * K + kk + c);
      }
      for (int t = threadIdx.x; t < 64 * 4; t += 256) {
        int r = t >> 2, c = (t & 3) * 8;
        *(uint4*)&sB[buf ^ 1][r * 32 + c] = *(const uint4*)(Bt + (long)(nblk + r) * K + kk + c);
      }
    }
#else
    __syncthreads();   // buffer reuse fence (loads for s+2 overwrite buf)
#endif
  }

  const int m0 = mblk + wm, n0 = nblk + wn;
  #pragma unroll
  for (int i = 0; i < 2; ++i) {
    #pragma unroll
    for (int j = 0; j < 2; ++j) {
      const int nn = n0 + j * 16 + lr;
      const float bv = bias ? bias[nn] : 0.f;
      #pragma unroll
      for (int jj = 0; jj < 8; ++jj) {
        const int mm = m0 + i * 16 + jj + 8 * hi;       // C layout: VGPR jj -> row
        float vsum = acc[i][j][jj] + bv;
        const long idx = (long)mm * N + nn;
        if (epi == 0) {
          ((u16*)out)[idx] = f2bf(vsum);
        } else if (epi == 1) {
          ((u16*)out)[idx] = f2bf(vsum + bf2f(addv[idx]));
        } else if (epi == 2) {
          float v2 = vsum + bf2f(addv[idx]);
          int bb = mm / S_LEN, ss = mm % S_LEN;
          ((u16*)out)[((long)bb * N + nn) * S_LEN + ss] = f2bf(v2);
        } else if (epi == 3) {
          ((float*)out)[idx] = vsum;
        } else {
          float g = 0.5f * vsum * (1.f + erff(vsum * 0.70710678118f));
          ((u16*)out)[idx] = f2bf(g);
        }
      }
    }
  }
}

// ------------------------------ attention ----------------------------------
// One block per (16-row q tile, batch). 8 waves.
// Phase1: scores(16x2048) = q @ kab^T * 1/8  -> bf16 in LDS.
// Phase2: rowwise softmax (normalization deferred to PV epilogue).
// Phase3: attn(16x1024) = P @ vpa, A-fragments from LDS, B from transposed vpa.
__global__ void attn_fused_kernel(const u16* __restrict__ q, const u16* __restrict__ kab,
                                  const u16* __restrict__ vpaT, u16* __restrict__ attnbf) {
  extern __shared__ u16 sS[];                       // [16][SPAD] + 16 float scales
  float* rowscale = (float*)(sS + 16 * SPAD);
  const int lane = threadIdx.x & 31;
  const int w    = threadIdx.x >> 5;
  const int lr   = lane & 15, hi = lane >> 4;
  const int q0   = blockIdx.x * 16;
  const int b    = blockIdx.y;
  const u16* qb = q   + ((long)b * S_LEN + q0) * H_DIM;
  const u16* kb = kab + (long)b * S_LEN * H_DIM;

  // --- phase 1: 16 x 2048 score strip; wave w owns columns [w*256, w*256+256)
  for (int nt = 0; nt < 16; ++nt) {
    const int n0 = w * 256 + nt * 16;
    v8f acc = {};
    for (int kk = 0; kk < H_DIM; kk += 32) {
      FragU a, bfr;
      frag_ld(a,   qb + (long)lr * H_DIM + kk + hi * 8);
      frag_ld(bfr, kb + (long)(n0 + lr) * H_DIM + kk + hi * 8);
      acc = __builtin_amdgcn_wmma_f32_16x16x32_bf16(
          false, a.v, false, bfr.v, (short)0, acc, false, false);
    }
    #pragma unroll
    for (int jj = 0; jj < 8; ++jj)
      sS[(jj + 8 * hi) * SPAD + n0 + lr] = f2bf(acc[jj] * 0.125f);  // 1/sqrt(64)
  }
  __syncthreads();

  // --- phase 2: softmax; wave w handles rows 2w, 2w+1
  for (int r = 2 * w; r < 2 * w + 2; ++r) {
    float mx = -1e30f;
    for (int c = lane; c < S_LEN; c += 32) mx = fmaxf(mx, bf2f(sS[r * SPAD + c]));
    #pragma unroll
    for (int off = 16; off; off >>= 1) mx = fmaxf(mx, __shfl_xor(mx, off));
    float sum = 0.f;
    for (int c = lane; c < S_LEN; c += 32) {
      float e = __expf(bf2f(sS[r * SPAD + c]) - mx);
      sS[r * SPAD + c] = f2bf(e);
      sum += e;
    }
    #pragma unroll
    for (int off = 16; off; off >>= 1) sum += __shfl_xor(sum, off);
    if (lane == 0) rowscale[r] = 1.f / sum;
  }
  __syncthreads();

  // --- phase 3: PV; wave w owns h columns [w*128, w*128+128)
  const int h0 = w * 128;
  v8f acc[8] = {};
  for (int kk = 0; kk < S_LEN; kk += 32) {
    FragU a;
    frag_ld(a, sS + (long)lr * SPAD + kk + hi * 8);          // ds_load_b128
    #pragma unroll
    for (int nt = 0; nt < 8; ++nt) {
      FragU bfr;
      frag_ld(bfr, vpaT + ((long)b * H_DIM + h0 + nt * 16 + lr) * S_LEN + kk + hi * 8);
      acc[nt] = __builtin_amdgcn_wmma_f32_16x16x32_bf16(
          false, a.v, false, bfr.v, (short)0, acc[nt], false, false);
    }
  }
  #pragma unroll
  for (int nt = 0; nt < 8; ++nt) {
    const int nn = h0 + nt * 16 + lr;
    #pragma unroll
    for (int jj = 0; jj < 8; ++jj) {
      const int mm = jj + 8 * hi;
      attnbf[((long)b * S_LEN + q0 + mm) * H_DIM + nn] = f2bf(acc[nt][jj] * rowscale[mm]);
    }
  }
}

// ------------------------- residual + layernorm ----------------------------
// One block (256 thr) per row of H=1024. Optional bf16 mirror for next GEMM.
__global__ void add_ln_kernel(const float* __restrict__ x, const float* __restrict__ res,
                              const float* __restrict__ gamma, const float* __restrict__ beta,
                              float* __restrict__ outf, u16* __restrict__ outbf) {
  __shared__ float redA[8], redB[8];
  const int row = blockIdx.x;
  const int lane = threadIdx.x & 31, w = threadIdx.x >> 5;
  const float* xr = x   + (long)row * H_DIM;
  const float* rr = res + (long)row * H_DIM;
  float v[4]; float s = 0.f;
  #pragma unroll
  for (int i = 0; i < 4; ++i) {
    v[i] = xr[threadIdx.x + i * 256] + rr[threadIdx.x + i * 256];
    s += v[i];
  }
  #pragma unroll
  for (int off = 16; off; off >>= 1) s += __shfl_xor(s, off);
  if (lane == 0) redA[w] = s;
  __syncthreads();
  float tot = 0.f;
  #pragma unroll
  for (int i = 0; i < 8; ++i) tot += redA[i];
  const float mu = tot * (1.f / H_DIM);
  float vs = 0.f;
  #pragma unroll
  for (int i = 0; i < 4; ++i) { float d = v[i] - mu; vs += d * d; }
  #pragma unroll
  for (int off = 16; off; off >>= 1) vs += __shfl_xor(vs, off);
  if (lane == 0) redB[w] = vs;
  __syncthreads();
  float tot2 = 0.f;
  #pragma unroll
  for (int i = 0; i < 8; ++i) tot2 += redB[i];
  const float rstd = rsqrtf(tot2 * (1.f / H_DIM) + 1e-5f);
  #pragma unroll
  for (int i = 0; i < 4; ++i) {
    const int c = threadIdx.x + i * 256;
    float y = (v[i] - mu) * rstd * gamma[c] + beta[c];
    outf[(long)row * H_DIM + c] = y;
    if (outbf) outbf[(long)row * H_DIM + c] = f2bf(y);
  }
}

// --------------------------------- host ------------------------------------
extern "C" void kernel_launch(void* const* d_in, const int* in_sizes, int n_in,
                              void* d_out, int out_size, void* d_ws, size_t ws_size,
                              hipStream_t stream) {
  const float* abs_enc = (const float*)d_in[0];
  const float* hidden  = (const float*)d_in[1];
  const float* Wq = (const float*)d_in[2];  const float* bq = (const float*)d_in[3];
  const float* Wk = (const float*)d_in[4];  const float* bk = (const float*)d_in[5];
  const float* Wv = (const float*)d_in[6];  const float* bv = (const float*)d_in[7];
  const float* W1 = (const float*)d_in[8];  const float* b1 = (const float*)d_in[9];
  const float* gamma = (const float*)d_in[10]; const float* beta = (const float*)d_in[11];
  const float* W2 = (const float*)d_in[12]; const float* b2 = (const float*)d_in[13];
  const float* W3 = (const float*)d_in[14]; const float* b3 = (const float*)d_in[15];

  char* p = (char*)d_ws;
  auto carve = [&](size_t bytes) { void* r = (void*)p; p += (bytes + 255) & ~(size_t)255; return r; };
  u16* hbf    = (u16*)carve((size_t)MROWS * H_DIM * 2);
  u16* abf    = (u16*)carve((size_t)MROWS * H_DIM * 2);
  u16* qbf    = (u16*)carve((size_t)MROWS * H_DIM * 2);
  u16* kabbf  = (u16*)carve((size_t)MROWS * H_DIM * 2);
  u16* vpaT   = (u16*)carve((size_t)MROWS * H_DIM * 2);   // [b][h][s]
  u16* attnbf = (u16*)carve((size_t)MROWS * H_DIM * 2);
  u16* a2bf   = (u16*)carve((size_t)MROWS * H_DIM * 2);
  u16* WqT = (u16*)carve((size_t)H_DIM * H_DIM * 2);
  u16* WkT = (u16*)carve((size_t)H_DIM * H_DIM * 2);
  u16* WvT = (u16*)carve((size_t)H_DIM * H_DIM * 2);
  u16* W1T = (u16*)carve((size_t)H_DIM * H_DIM * 2);
  u16* W2T = (u16*)carve((size_t)H_DIM * I_DIM * 2);
  u16* W3T = (u16*)carve((size_t)I_DIM * H_DIM * 2);
  u16* ff1bf  = (u16*)carve((size_t)MROWS * I_DIM * 2);
  float* t1f  = (float*)carve((size_t)MROWS * H_DIM * 4);
  float* a2f  = (float*)carve((size_t)MROWS * H_DIM * 4);
  float* ff2f = (float*)carve((size_t)MROWS * H_DIM * 4);

  const long nH = (long)MROWS * H_DIM;
  cvt_bf16_kernel<<<dim3((unsigned)((nH + 255) / 256)), 256, 0, stream>>>(hidden,  hbf, nH);
  cvt_bf16_kernel<<<dim3((unsigned)((nH + 255) / 256)), 256, 0, stream>>>(abs_enc, abf, nH);

  const long wHH = (long)H_DIM * H_DIM, wHI = (long)H_DIM * I_DIM;
  transpose_cvt_kernel<<<dim3((unsigned)((wHH + 255) / 256)), 256, 0, stream>>>(Wq, WqT, H_DIM, H_DIM);
  transpose_cvt_kernel<<<dim3((unsigned)((wHH + 255) / 256)), 256, 0, stream>>>(Wk, WkT, H_DIM, H_DIM);
  transpose_cvt_kernel<<<dim3((unsigned)((wHH + 255) / 256)), 256, 0, stream>>>(Wv, WvT, H_DIM, H_DIM);
  transpose_cvt_kernel<<<dim3((unsigned)((wHH + 255) / 256)), 256, 0, stream>>>(W1, W1T, H_DIM, H_DIM);
  transpose_cvt_kernel<<<dim3((unsigned)((wHI + 255) / 256)), 256, 0, stream>>>(W2, W2T, H_DIM, I_DIM);
  transpose_cvt_kernel<<<dim3((unsigned)((wHI + 255) / 256)), 256, 0, stream>>>(W3, W3T, I_DIM, H_DIM);

  dim3 gHH(MROWS / 128, H_DIM / 64);
  // q = h@Wq+bq ; kab = h@Wk+bk+abs ; vpa^T = (h@Wv+bv+abs)^T
  gemm_bf16_wmma<<<gHH, 256, 0, stream>>>(hbf, WqT, bq, nullptr, qbf,   MROWS, H_DIM, H_DIM, 0);
  gemm_bf16_wmma<<<gHH, 256, 0, stream>>>(hbf, WkT, bk, abf,     kabbf, MROWS, H_DIM, H_DIM, 1);
  gemm_bf16_wmma<<<gHH, 256, 0, stream>>>(hbf, WvT, bv, abf,     vpaT,  MROWS, H_DIM, H_DIM, 2);

  const size_t smem = (size_t)16 * SPAD * 2 + 16 * sizeof(float);
  attn_fused_kernel<<<dim3(S_LEN / 16, B_SZ), 256, smem, stream>>>(qbf, kabbf, vpaT, attnbf);

  // t1 = attn@W1+b1 (f32) ; attn2 = LN(t1 + hidden)
  gemm_bf16_wmma<<<gHH, 256, 0, stream>>>(attnbf, W1T, b1, nullptr, t1f, MROWS, H_DIM, H_DIM, 3);
  add_ln_kernel<<<dim3(MROWS), 256, 0, stream>>>(t1f, hidden, gamma, beta, a2f, a2bf);

  // ff = gelu(attn2@W2+b2) @ W3 + b3 ; out = LN(ff + attn2)
  gemm_bf16_wmma<<<dim3(MROWS / 128, I_DIM / 64), 256, 0, stream>>>(a2bf, W2T, b2, nullptr, ff1bf, MROWS, I_DIM, H_DIM, 4);
  gemm_bf16_wmma<<<gHH, 256, 0, stream>>>(ff1bf, W3T, b3, nullptr, ff2f, MROWS, H_DIM, I_DIM, 3);
  add_ln_kernel<<<dim3(MROWS), 256, 0, stream>>>(ff2f, a2f, gamma, beta, (float*)d_out, nullptr);
}
